// DeltaNetCell_48378511622733
// MI455X (gfx1250) — matrix operand order repaired and checked by reading, not compiled, not run on hardware
//
#include <hip/hip_runtime.h>
#include <math.h>

// Problem constants (match reference)
#define B_    4
#define S_    4096
#define INDIM 1024
#define HID_  1024
#define NH_   16
#define CS_   16
#define HD_   64
#define NC_   (S_ / CS_)      // 256 chunks
#define TOK   (B_ * S_)      // 16384 tokens
#define KDIM  1024            // K of both big GEMMs
#define NDIM  1024            // N of both big GEMMs

typedef __bf16 v16bf  __attribute__((ext_vector_type(16)));
typedef float  v8f    __attribute__((ext_vector_type(8)));
typedef __bf16 bf16x4 __attribute__((ext_vector_type(4)));

// Types for the async-copy builtin: first param is printed by clang as
// 'int __vector(4) __device__ *'  ==  vector_size(16) int, AS1 ("__device__").
typedef int v4i_ld __attribute__((vector_size(16)));
typedef v4i_ld __attribute__((address_space(1)))* gv4p;   // global (AS1)
typedef v4i_ld __attribute__((address_space(3)))* lv4p;   // LDS (AS3)

__device__ __forceinline__ __bf16 to_bf16(float f) {
  unsigned u = __builtin_bit_cast(unsigned, f);
  unsigned r = u + 0x7FFFu + ((u >> 16) & 1u);   // round-to-nearest-even
  unsigned short h = (unsigned short)(r >> 16);
  return __builtin_bit_cast(__bf16, h);
}

__device__ __forceinline__ float sigmoidf_(float x) { return 1.0f / (1.0f + expf(-x)); }

// ---- CDNA5 async global->LDS copy (ASYNCcnt-tracked), with safe fallback ----
#if __has_builtin(__builtin_amdgcn_global_load_async_to_lds_b128)
#define HAVE_ASYNC 1
#else
#define HAVE_ASYNC 0
#endif

__device__ __forceinline__ void cp_async16(const __bf16* g, __bf16* l) {
#if HAVE_ASYNC
  __builtin_amdgcn_global_load_async_to_lds_b128((gv4p)g, (lv4p)l, 0, 0);
#else
  *(uint4*)l = *(const uint4*)g;
#endif
}

__device__ __forceinline__ void cp_wait_all() {
#if HAVE_ASYNC
#if __has_builtin(__builtin_amdgcn_s_wait_asynccnt)
  __builtin_amdgcn_s_wait_asynccnt(0);
#else
  asm volatile("s_wait_asynccnt 0x0" ::: "memory");
#endif
#endif
}

// ---------------------------------------------------------------------------
// Pre-pass A: f32 -> bf16 elementwise (for x)
// ---------------------------------------------------------------------------
__global__ __launch_bounds__(256)
void convert_f32_to_bf16(const float* __restrict__ in, __bf16* __restrict__ outp)
{
  size_t i = (size_t)blockIdx.x * 256 + threadIdx.x;   // 4-element groups
  float4 v = ((const float4*)in)[i];
  bf16x4 o;
  o[0] = to_bf16(v.x); o[1] = to_bf16(v.y); o[2] = to_bf16(v.z); o[3] = to_bf16(v.w);
  ((bf16x4*)outp)[i] = o;
}

// ---------------------------------------------------------------------------
// Pre-pass B: W[n][k] f32 -> Wt[k][n] bf16 (k-major weights for B fragments)
// ---------------------------------------------------------------------------
__global__ __launch_bounds__(256)
void transpose_f32_to_bf16(const float* __restrict__ W, __bf16* __restrict__ Wt)
{
  __shared__ __bf16 tile[32][33];
  const int bx = blockIdx.x & 31;    // k tile
  const int by = blockIdx.x >> 5;    // n tile
  const int tx = threadIdx.x & 31, ty = threadIdx.x >> 5;
  #pragma unroll
  for (int q = 0; q < 4; ++q) {
    int n = by * 32 + ty + q * 8;
    tile[ty + q * 8][tx] = to_bf16(W[(size_t)n * KDIM + bx * 32 + tx]);
  }
  __syncthreads();
  #pragma unroll
  for (int q = 0; q < 4; ++q) {
    int k = bx * 32 + ty + q * 8;
    Wt[(size_t)k * NDIM + by * 32 + tx] = tile[tx][ty + q * 8];
  }
}

// ---------------------------------------------------------------------------
// Big GEMM: C[16384,1024] = A(bf16, MxK row-major) @ Bt(bf16, KxN k-major)
// 128x64 block tile, 8 waves, 2x2 WMMA per wave, double-buffered async LDS.
// ---------------------------------------------------------------------------
#define TM 128
#define TN 64
#define TK 32
#define APAD 8
#define BPAD 8
#define KITERS (KDIM / TK)   // 32

__global__ __launch_bounds__(256, 2)
void gemm_bf16_wmma(const __bf16* __restrict__ A, const __bf16* __restrict__ Bt,
                    float* __restrict__ C)
{
  __shared__ __bf16 As[2][TM][TK + APAD];   // 20 KiB
  __shared__ __bf16 Bs[2][TK][TN + BPAD];   //  9 KiB

  const int tid  = threadIdx.x;
  const int lane = tid & 31;
  const int wave = tid >> 5;
  const int m0 = blockIdx.x * TM;
  const int n0 = blockIdx.y * TN;
  const int wm = (wave >> 1) * 32;
  const int wn = (wave & 1) * 32;

  auto stage = [&](int buf, int t) {
    const int k0 = t * TK;
    #pragma unroll
    for (int it = 0; it < 2; ++it) {            // A tile: 128x32 halves
      int r = (tid >> 2) + it * 64;
      int c = (tid & 3) << 3;
      cp_async16(A + (size_t)(m0 + r) * KDIM + k0 + c, &As[buf][r][c]);
    }
    {                                           // B tile: 32x64 halves (k-major)
      int r = tid >> 3;
      int c = (tid & 7) << 3;
      cp_async16(Bt + (size_t)(k0 + r) * NDIM + n0 + c, &Bs[buf][r][c]);
    }
  };

  v8f acc[2][2] = {};
  stage(0, 0);
  cp_wait_all();
  __syncthreads();

  for (int t = 0; t < KITERS; ++t) {
    const int cur = t & 1;
    if (t + 1 < KITERS) stage(cur ^ 1, t + 1);
    if (t + 2 < KITERS) {                       // prime L2 for tile t+2
      const int k2 = (t + 2) * TK;
      __builtin_prefetch(A + (size_t)(m0 + (tid >> 2)) * KDIM + k2 + ((tid & 3) << 3), 0, 3);
      __builtin_prefetch(Bt + (size_t)(k2 + (tid >> 3)) * NDIM + n0 + ((tid & 7) << 3), 0, 3);
    }

    // A fragment (16-bit 16x32): row = lane&15; lanes 0-15: K 0..7/16..23,
    // lanes 16-31: K 8..15/24..31.  B fragment (32x16): N = lane&15, elem = K (same K order as A).
    v16bf af[2], bfg[2];
    const int arow = lane & 15;
    const int kb   = (lane >> 4) << 3;
    #pragma unroll
    for (int im = 0; im < 2; ++im) {
      #pragma unroll
      for (int e = 0; e < 8; ++e)  af[im][e] = As[cur][wm + im * 16 + arow][kb + e];
      #pragma unroll
      for (int e = 8; e < 16; ++e) af[im][e] = As[cur][wm + im * 16 + arow][kb + e + 8];
    }
    #pragma unroll
    for (int in = 0; in < 2; ++in) {
      #pragma unroll
      for (int e = 0; e < 8; ++e)  bfg[in][e] = Bs[cur][kb + e][wn + in * 16 + arow];
      #pragma unroll
      for (int e = 8; e < 16; ++e) bfg[in][e] = Bs[cur][kb + e + 8][wn + in * 16 + arow];
    }

    #pragma unroll
    for (int im = 0; im < 2; ++im)
      #pragma unroll
      for (int in = 0; in < 2; ++in)
        acc[im][in] = __builtin_amdgcn_wmma_f32_16x16x32_bf16(
            false, af[im], false, bfg[in], (short)0, acc[im][in], false, false);

    if (t + 1 < KITERS) { cp_wait_all(); __syncthreads(); }
  }

  // C layout: VGPR r -> M=r (lanes 0-15) / M=r+8 (lanes 16-31); N = lane&15
  const int mh = (lane >> 4) << 3;
  #pragma unroll
  for (int im = 0; im < 2; ++im)
    #pragma unroll
    for (int in = 0; in < 2; ++in) {
      float* cp = C + (size_t)(m0 + wm + im * 16 + mh) * NDIM
                    + (n0 + wn + in * 16 + (lane & 15));
      #pragma unroll
      for (int r = 0; r < 8; ++r)
        cp[(size_t)r * NDIM] = acc[im][in][r];   // constant offsets -> imm ioffset
    }
}

// ---------------------------------------------------------------------------
// Per-token LayerNorm + gates + forget + a / b_in production
// ---------------------------------------------------------------------------
__global__ __launch_bounds__(256)
void fused_ln_gates(const float* __restrict__ x, const float* __restrict__ y_raw,
                    const float* __restrict__ ln_g, const float* __restrict__ ln_b,
                    const float* __restrict__ W_gate, const float* __restrict__ b_gate,
                    const float* __restrict__ W_forget, const float* __restrict__ eig_raw,
                    float* __restrict__ a_buf, float* __restrict__ b_in)
{
  __shared__ float xp[HID_];
  __shared__ float xr[INDIM];
  __shared__ float red[256], red2[256];
  __shared__ float gout[NH_ * 3];
  __shared__ float coef[NH_];
  __shared__ float s_mu, s_rstd;

  const int t = blockIdx.x;
  const int tid = threadIdx.x;

  float s = 0.f, s2 = 0.f;
  #pragma unroll
  for (int q = 0; q < 4; ++q) {
    int j = tid + q * 256;
    float yv = y_raw[(size_t)t * HID_ + j];
    xp[j] = yv;
    xr[j] = x[(size_t)t * INDIM + j];
    s += yv; s2 += yv * yv;
  }
  red[tid] = s; red2[tid] = s2;
  __syncthreads();
  for (int off = 128; off > 0; off >>= 1) {
    if (tid < off) { red[tid] += red[tid + off]; red2[tid] += red2[tid + off]; }
    __syncthreads();
  }
  if (tid == 0) {
    float mu  = red[0] * (1.f / HID_);
    float var = red2[0] * (1.f / HID_) - mu * mu;
    s_mu = mu; s_rstd = rsqrtf(var + 1e-5f);
  }
  __syncthreads();
  const float mu = s_mu, rstd = s_rstd;
  #pragma unroll
  for (int q = 0; q < 4; ++q) {
    int j = tid + q * 256;
    xp[j] = (xp[j] - mu) * rstd * ln_g[j] + ln_b[j];
  }
  __syncthreads();

  const int lane = tid & 31, wave = tid >> 5;
  for (int o = wave; o < NH_ * 3; o += 8) {
    const float* wrow;
    const float* vec;
    if (o < 2 * NH_) { wrow = W_gate + (size_t)o * HID_;                vec = xp; }
    else             { wrow = W_forget + (size_t)(o - 2 * NH_) * INDIM; vec = xr; }
    float d = 0.f;
    for (int j = lane; j < HID_; j += 32) d += vec[j] * wrow[j];
    #pragma unroll
    for (int off = 16; off > 0; off >>= 1) d += __shfl_xor(d, off, 32);
    if (lane == 0) gout[o] = d;
  }
  __syncthreads();

  if (tid < NH_) {
    int hh = tid;
    float alpha  = sigmoidf_(gout[hh] + b_gate[hh]);
    float beta_g = sigmoidf_(gout[NH_ + hh] + b_gate[NH_ + hh]);
    float fg     = sigmoidf_(gout[2 * NH_ + hh]);
    float eig    = tanhf(eig_raw[hh]);
    a_buf[(size_t)t * NH_ + hh] = fg * eig * alpha;
    coef[hh] = (1.f - fg) * beta_g;
  }
  __syncthreads();
  #pragma unroll
  for (int q = 0; q < 4; ++q) {
    int j = tid + q * 256;
    b_in[(size_t)t * HID_ + j] = coef[j >> 6] * xp[j];
  }
}

// ---------------------------------------------------------------------------
// Per (b,chunk,head) wave: decay matrix M, cum_a, contrib = M @ b_c (bf16 WMMA)
// ---------------------------------------------------------------------------
__global__ __launch_bounds__(128)
void chunk_contrib(const float* __restrict__ a_buf, const float* __restrict__ b_in,
                   float* __restrict__ cum_a, float* __restrict__ contrib)
{
  __shared__ __bf16 Ms[4][16][TK];   // cols 16..31 zero padding
  __shared__ float lc_s[4][16], sp_s[4][16];

  const int bc = blockIdx.x;
  const int b = bc / NC_, c = bc % NC_;
  const int tid = threadIdx.x;
  const int lane = tid & 31, wave = tid >> 5;
  const int tok0 = b * S_ + c * CS_;

  for (int hh = 0; hh < 4; ++hh) {
    const int h = wave * 4 + hh;

    float lc = 0.f, sg = 1.f;
    if (lane < 16) {
      float av = a_buf[(size_t)(tok0 + lane) * NH_ + h];
      sg = (av < 0.f) ? -1.f : 1.f;
      lc = logf(fmaxf(fabsf(av), 1e-7f));
    }
    #pragma unroll
    for (int off = 1; off < 16; off <<= 1) {
      float tl = __shfl_up(lc, off, 32);
      float ts = __shfl_up(sg, off, 32);
      if (lane >= off && lane < 16) { lc += tl; sg *= ts; }
    }
    if (lane < 16) {
      cum_a[(size_t)(tok0 + lane) * NH_ + h] = sg * expf(lc);
      lc_s[wave][lane] = lc;
      sp_s[wave][lane] = sg;
    }
    __syncthreads();

    #pragma unroll
    for (int q = 0; q < 16; ++q) {
      int f = lane * 16 + q;
      int i = f >> 5, col = f & 31;
      float v = 0.f;
      if (col < 16) {
        if (col == i) v = 1.f;
        else if (col < i) {
          float lcp = (col == 0) ? 0.f : lc_s[wave][col - 1];
          float spp = (col == 0) ? 1.f : sp_s[wave][col - 1];
          v = expf(lc_s[wave][i] - lcp) * sp_s[wave][i] * spp;
        }
      }
      Ms[wave][i][col] = to_bf16(v);
    }
    __syncthreads();

    v16bf af;
    const int arow = lane & 15;
    const int kb = (lane >> 4) << 3;
    #pragma unroll
    for (int e = 0; e < 8; ++e)  af[e] = Ms[wave][arow][kb + e];
    #pragma unroll
    for (int e = 8; e < 16; ++e) af[e] = Ms[wave][arow][kb + e + 8];

    #pragma unroll
    for (int nt = 0; nt < 4; ++nt) {
      v16bf bfg;
      {
        const float* src = b_in + (size_t)(tok0 + kb) * HID_ + h * HD_ + nt * 16 + arow;
        #pragma unroll
        for (int e = 0; e < 8; ++e) bfg[e] = to_bf16(src[(size_t)e * HID_]);
        #pragma unroll
        for (int e = 8; e < 16; ++e) bfg[e] = to_bf16(0.f);
      }
      v8f accv = {};
      accv = __builtin_amdgcn_wmma_f32_16x16x32_bf16(
          false, af, false, bfg, (short)0, accv, false, false);

      const int mh = (lane >> 4) << 3;
      const int n  = nt * 16 + (lane & 15);
      #pragma unroll
      for (int r = 0; r < 8; ++r)
        contrib[(size_t)(tok0 + r + mh) * HID_ + h * HD_ + n] = accv[r];
    }
    __syncthreads();
  }
}

// ---------------------------------------------------------------------------
// 256-step chunk-carry scan; writes h_all directly as bf16 for the W_out GEMM
// ---------------------------------------------------------------------------
__global__ __launch_bounds__(1024)
void chunk_scan(const float* __restrict__ h0, const float* __restrict__ cum_a,
                const float* __restrict__ contrib, __bf16* __restrict__ h_all_bf,
                float* __restrict__ h_final)
{
  __shared__ float hprev[HD_];
  const int b  = blockIdx.x >> 4;
  const int hh = blockIdx.x & 15;
  const int tid = threadIdx.x;
  const int i = tid >> 6;
  const int d = tid & 63;

  if (tid < HD_) hprev[tid] = h0[(size_t)b * HID_ + hh * HD_ + tid];
  __syncthreads();

  for (int c = 0; c < NC_; ++c) {
    const int tok = b * S_ + c * CS_ + i;
    float ca = cum_a[(size_t)tok * NH_ + hh];
    float hv = fmaf(ca, hprev[d], contrib[(size_t)tok * HID_ + hh * HD_ + d]);
    h_all_bf[(size_t)tok * HID_ + hh * HD_ + d] = to_bf16(hv);
    __syncthreads();
    if (i == CS_ - 1) hprev[d] = hv;
    __syncthreads();
  }
  if (i == CS_ - 1) h_final[(size_t)b * HID_ + hh * HD_ + d] = hprev[d];
}

// ---------------------------------------------------------------------------
extern "C" void kernel_launch(void* const* d_in, const int* in_sizes, int n_in,
                              void* d_out, int out_size, void* d_ws, size_t ws_size,
                              hipStream_t stream)
{
  const float* x        = (const float*)d_in[0];
  const float* h0       = (const float*)d_in[1];
  const float* W_in     = (const float*)d_in[2];
  const float* ln_g     = (const float*)d_in[3];
  const float* ln_b     = (const float*)d_in[4];
  const float* W_gate   = (const float*)d_in[5];
  const float* b_gate   = (const float*)d_in[6];
  const float* W_forget = (const float*)d_in[7];
  const float* W_out    = (const float*)d_in[8];
  const float* eig_raw  = (const float*)d_in[9];

  float* out     = (float*)d_out;
  float* h_final = out + (size_t)TOK * HID_;

  // Workspace carve-up (~230 MB).
  char* ws = (char*)d_ws;
  const size_t big = (size_t)TOK * HID_ * sizeof(float);   // 64 MiB
  float*  y_raw   = (float*)(ws);                          // dead after fused_ln
  float*  b_in    = (float*)(ws + big);                    // dead after chunk_contrib
  float*  contrib = (float*)(ws + 2 * big);
  float*  a_buf   = (float*)(ws + 3 * big);
  float*  cum_a   = (float*)(ws + 3 * big + (size_t)TOK * NH_ * sizeof(float));
  __bf16* xbf     = (__bf16*)(ws + 3 * big + 2 * (size_t)TOK * NH_ * sizeof(float));
  __bf16* Wt_in   = (__bf16*)((char*)xbf + (size_t)TOK * HID_ * sizeof(__bf16));
  __bf16* Wt_out  = (__bf16*)((char*)Wt_in + (size_t)NDIM * KDIM * sizeof(__bf16));
  __bf16* hbf     = (__bf16*)b_in;                         // aliases dead b_in

  dim3 gdim(TOK / TM, NDIM / TN);   // (128, 16)

  convert_f32_to_bf16<<<TOK * HID_ / 1024, 256, 0, stream>>>(x, xbf);
  transpose_f32_to_bf16<<<1024, 256, 0, stream>>>(W_in, Wt_in);
  transpose_f32_to_bf16<<<1024, 256, 0, stream>>>(W_out, Wt_out);

  gemm_bf16_wmma<<<gdim, 256, 0, stream>>>(xbf, Wt_in, y_raw);
  fused_ln_gates<<<TOK, 256, 0, stream>>>(x, y_raw, ln_g, ln_b, W_gate, b_gate,
                                          W_forget, eig_raw, a_buf, b_in);
  chunk_contrib<<<B_ * NC_, 128, 0, stream>>>(a_buf, b_in, cum_a, contrib);
  chunk_scan<<<B_ * NH_, 1024, 0, stream>>>(h0, cum_a, contrib, hbf, h_final);
  gemm_bf16_wmma<<<gdim, 256, 0, stream>>>(hbf, Wt_out, out);
}
